// CoarseEdgeCoarsenAndFuse_23819888623885
// MI455X (gfx1250) — matrix operand order repaired and checked by reading, not compiled, not run on hardware
//
#include <hip/hip_runtime.h>
#include <hip/hip_bf16.h>

// ---------------------------------------------------------------------------
// CoarseEdgeCoarsenAndFuse for MI455X (gfx1250, wave32, WMMA).
//
// Precision: fp32 inputs -> bf16 GEMM operands, fp32 accumulate via
// v_wmma_f32_16x16x32_bf16. Epilogues (scaling, LN, SiLU, RBF, masks) fp32.
//
// Roofline: ~0.5 GB mandatory HBM traffic (~21 us @ 23.3 TB/s) vs ~50 GFLOP;
// bf16 WMMA keeps the GEMMs at/below memory time.
//
// GEMM1 pipeline (this revision):
//  - Zb scratch is stored in global memory ALREADY in WMMA fragment-tile
//    order (32x16 tile = 512 contiguous halves, lane-major), produced by the
//    memory-bound conversion kernel k_zbf_swz via an LDS transpose.
//  - GEMM1 staging is therefore pure GLOBAL_LOAD_ASYNC_TO_LDS_B128:
//    1 async copy/thread for A, 2 for B; zero VGPR round trips, zero
//    ds_stores, zero packing VALU in the hot loop.
//  - double-buffered LDS, s_wait_asynccnt + one barrier per k-step
//  - fragments: A = 2x ds_load_b128, B = 2x ds_load_b128 (32B/lane)
// ---------------------------------------------------------------------------

typedef __attribute__((ext_vector_type(8)))  float  v8f;
typedef __attribute__((ext_vector_type(16))) __bf16 v16bf;

constexpr int   B_    = 2;
constexpr int   N_    = 384;
constexpr int   K_    = 384;
constexpr int   CZ_   = 128;
constexpr int   COUT_ = 128;
constexpr int   BINS_ = 16;
constexpr int   CIN_  = CZ_ + BINS_;   // 144
constexpr int   CINP_ = 160;           // padded GEMM K (multiple of 32)
constexpr int   NCOL_ = N_ * CZ_;      // 49152 flattened (m,z)
constexpr int   NCT_  = NCOL_ / 16;    // 3072 col tiles per (b, ktile)
constexpr int   NKT_  = N_ / 32;       // 12 k tiles
constexpr int   TSTR_ = 528;           // LDS B tile stride in halves (1056B, 32B-aligned)
constexpr float EPS_    = 1e-8f;
constexpr float LN_EPS_ = 1e-5f;
constexpr float DMAX_   = 10.0f;

// ---- WMMA fragment loaders -------------------------------------------------
// A: 16x32 bf16, row-major LDS tile with padded stride `ld`.
// lanes 0-15: M=lane, halves 0-15 -> K=0..15
// lanes 16-31: M=lane-16, K=16..31. Contiguous 8+8 -> 2x ds_load_b128.
__device__ __forceinline__ v16bf lds_a_16x32(const __bf16* p, int ld, int lane) {
  const int m  = lane & 15;
  const int kb = (lane & 16) ? 16 : 0;
  const __bf16* r = p + m * ld;
  v16bf a;
#pragma unroll
  for (int e = 0; e < 8; ++e) a[e] = r[kb + e];
#pragma unroll
  for (int e = 0; e < 8; ++e) a[8 + e] = r[kb + 8 + e];
  return a;
}
// B: 32x16 bf16 tile in fragment order: lane l's 16 halves contiguous at
// tile + l*16. One aligned 32B read per fragment.
__device__ __forceinline__ v16bf lds_b_swz(const __bf16* tile, int lane) {
  return *reinterpret_cast<const v16bf*>(tile + lane * 16);
}
// Fragment-order index for element (kk in 0..31, nn in 0..15) of one tile:
// lane = nn + (kk & 16), element = kk & 15.
__device__ __forceinline__ int b_swz_idx(int kk, int nn) {
  return (nn + (kk & 16)) * 16 + (kk & 15);
}
// 16B global -> LDS async copy (ASYNCcnt-tracked, no VGPR round trip).
__device__ __forceinline__ void async_cp16(void* lds, const void* gsrc) {
  unsigned ldso = (unsigned)(size_t)lds;
  asm volatile("global_load_async_to_lds_b128 %0, %1, off"
               :: "v"(ldso), "v"(gsrc) : "memory");
}

// ---- small prep kernels ----------------------------------------------------
__global__ void k_rowsum(const float* __restrict__ A, const float* __restrict__ mf,
                         float* __restrict__ s) {
  int gid = blockIdx.x * blockDim.x + threadIdx.x;
  if (gid >= B_ * N_) return;
  const float* row = A + (size_t)gid * K_;
  float acc = 0.f;
  for (int k = 0; k < K_; ++k) acc += row[k];
  s[gid] = acc * mf[gid];
}

__global__ void k_colsum(const float* __restrict__ A, const float* __restrict__ mf,
                         float* __restrict__ col) {
  int gid = blockIdx.x * blockDim.x + threadIdx.x;
  if (gid >= B_ * K_) return;
  int b = gid / K_, k = gid % K_;
  float acc = 0.f;
  for (int n = 0; n < N_; ++n) acc += A[((size_t)b * N_ + n) * K_ + k] * mf[b * N_ + n];
  col[gid] = acc;
}

// AT[b][a][n] = A[b][n][a] * mf[b][n]  (bf16), 32x32 LDS transpose tiles.
__global__ __launch_bounds__(256) void k_at(const float* __restrict__ A,
                                            const float* __restrict__ mf,
                                            __bf16* __restrict__ AT) {
  __shared__ float tile[32][33];
  const int b = blockIdx.z, n0 = blockIdx.y * 32, a0 = blockIdx.x * 32;
  const int tx = threadIdx.x & 31, ty = threadIdx.x >> 5;
#pragma unroll
  for (int p = 0; p < 4; ++p) {
    int n = n0 + ty + p * 8;
    tile[ty + p * 8][tx] = A[((size_t)b * N_ + n) * K_ + a0 + tx] * mf[b * N_ + n];
  }
  __syncthreads();
#pragma unroll
  for (int p = 0; p < 4; ++p) {
    int ar = ty + p * 8;
    AT[((size_t)b * K_ + a0 + ar) * N_ + n0 + tx] = (__bf16)tile[tx][ar];
  }
}

// ZbS = bf16(Z_fine * mf[n]*mf[m]) rearranged to fragment-tile order:
//   ZbS[b][kt(12)][ct(3072)][512], tile = 32(k=n) x 16(col), lane-major.
// Block: 32 n-rows x 128 mz-cols; coalesced f32 reads, LDS scatter, then
// perfectly coalesced contiguous 8KB chunk write (32B per thread).
__global__ __launch_bounds__(256) void k_zbf_swz(const float* __restrict__ Z,
                                                 const float* __restrict__ mf,
                                                 __bf16* __restrict__ ZbS) {
  __shared__ __align__(32) __bf16 st[8 * 512];  // 8KB, 8 fragment-order tiles
  const int b = blockIdx.z, kt = blockIdx.y, cg = blockIdx.x;
  const int t = threadIdx.x;
  const int r = t >> 3, c0 = (t & 7) * 16;      // row 0..31, col base 0..112
  const int n = kt * 32 + r;
  const float mm = mf[b * N_ + n] * mf[b * N_ + cg];  // m == cg (128-aligned cols)
  const float* src = Z + ((size_t)b * N_ + n) * NCOL_ + (size_t)cg * 128 + c0;
  const int e = r & 15, lh = r & 16;            // kk == r
#pragma unroll
  for (int q = 0; q < 4; ++q) {
    float4 v = *reinterpret_cast<const float4*>(src + q * 4);
    float vv[4] = {v.x, v.y, v.z, v.w};
#pragma unroll
    for (int u = 0; u < 4; ++u) {
      int c = c0 + q * 4 + u;
      st[(c >> 4) * 512 + ((c & 15) + lh) * 16 + e] = (__bf16)(vv[u] * mm);
    }
  }
  __syncthreads();
  const size_t outBase = (((size_t)b * NKT_ + kt) * NCT_ + (size_t)cg * 8) * 512;
  const uint4* s = reinterpret_cast<const uint4*>(st) + t * 2;
  uint4* d = reinterpret_cast<uint4*>(ZbS + outBase) + t * 2;
  d[0] = s[0];
  d[1] = s[1];
}

// ---- GEMM1: Z_sem_c[b,a,(m,z)] = (AT x Zb) * s[m] / max(col[a]*s[m], eps) --
__global__ __launch_bounds__(256) void gemm1_sem(const __bf16* __restrict__ AT,
                                                 const __bf16* __restrict__ ZbS,
                                                 const float* __restrict__ sbuf,
                                                 const float* __restrict__ colbuf,
                                                 float* __restrict__ out_sem) {
  __shared__ __align__(32) __bf16 As[2][64 * 40];      // 64 rows x 32 K (+8 pad)
  __shared__ __align__(32) __bf16 Bsw[2][8 * TSTR_];   // 8 fragment-order tiles
  const int t = threadIdx.x, lane = t & 31, w = t >> 5;
  const int b = blockIdx.z;
  const int rowBase = blockIdx.y * 64;
  const int colBase = blockIdx.x * 128;
  const int m = colBase >> 7;          // CZ==128 && 128-aligned tile -> fixed m
  const int wm = w >> 1, wn = w & 1;   // 4 row-waves x 2 col-waves
  v8f acc[4];
#pragma unroll
  for (int j = 0; j < 4; ++j) acc[j] = {};

  const __bf16* Ag  = AT + ((size_t)b * K_ + rowBase) * N_;
  const __bf16* BgS = ZbS + ((size_t)b * NKT_ * NCT_ + (size_t)blockIdx.x * 8) * 512;
  const int ar = t >> 2, ac = (t & 3) * 8;   // A stage: 64 rows x 32
  const int bj = t >> 5, bo = (t & 31) * 16; // B stage: 8 tiles x 32 chunks

  auto stageA = [&](int buf, int kt) {       // 1 async 16B copy / thread
    async_cp16(&As[buf][ar * 40 + ac], Ag + (size_t)ar * N_ + kt * 32 + ac);
  };
  auto stageB = [&](int buf, int kt) {       // 2 async 16B copies / thread
    const __bf16* src = BgS + (size_t)kt * NCT_ * 512 + (size_t)bj * 512 + bo;
    __bf16* dst = &Bsw[buf][bj * TSTR_ + bo];
    async_cp16(dst, src);
    async_cp16(dst + 8, src + 8);
  };

  stageA(0, 0);
  stageB(0, 0);
  for (int kt = 0; kt < NKT_; ++kt) {
    const int cur = kt & 1;
    asm volatile("s_wait_asynccnt 0x0" ::: "memory");  // tile `cur` landed
    __syncthreads();                                   // publish tile `cur`
    if (kt + 1 < NKT_) {                               // overlap next stage
      stageA(cur ^ 1, kt + 1);
      stageB(cur ^ 1, kt + 1);
    }
    if (kt + 2 < NKT_) {                               // global_prefetch_b8
      __builtin_prefetch(Ag + (size_t)ar * N_ + (kt + 2) * 32 + ac, 0, 1);
      __builtin_prefetch(BgS + (size_t)(kt + 2) * NCT_ * 512 + bj * 512 + bo, 0, 1);
    }
    v16bf af = lds_a_16x32(&As[cur][(wm * 16) * 40], 40, lane);
#pragma unroll
    for (int j = 0; j < 4; ++j) {
      v16bf bf = lds_b_swz(&Bsw[cur][(wn * 4 + j) * TSTR_], lane);
      acc[j] = __builtin_amdgcn_wmma_f32_16x16x32_bf16(false, af, false, bf,
                                                       (short)0, acc[j], false, false);
    }
  }

  // epilogue: scale = s_m / max(col_a * s_m, eps)
  const float s_m = sbuf[b * N_ + m];
  const int mrow = (lane & 16) ? 8 : 0;
  float scale[8];
#pragma unroll
  for (int v = 0; v < 8; ++v) {
    int a_ = rowBase + wm * 16 + v + mrow;
    scale[v] = s_m / fmaxf(colbuf[b * K_ + a_] * s_m, EPS_);
  }
#pragma unroll
  for (int j = 0; j < 4; ++j) {
    int cl = colBase + wn * 64 + j * 16 + (lane & 15);
#pragma unroll
    for (int v = 0; v < 8; ++v) {
      int a_ = rowBase + wm * 16 + v + mrow;
      out_sem[((size_t)b * K_ + a_) * NCOL_ + cl] = acc[j][v] * scale[v];
    }
  }
}

// ---- RBF geometry branch ---------------------------------------------------
__global__ void k_rbf(const float* __restrict__ mu, const float* __restrict__ mc,
                      float* __restrict__ geo) {
  const float width = DMAX_ / (BINS_ - 1);
  int gid = blockIdx.x * blockDim.x + threadIdx.x;
  if (gid >= B_ * K_ * K_) return;
  int b = gid / (K_ * K_), rem = gid % (K_ * K_);
  int i = rem / K_, j = rem % K_;
  const float* pi = mu + ((size_t)b * K_ + i) * 3;
  const float* pj = mu + ((size_t)b * K_ + j) * 3;
  float dx = pi[0] - pj[0], dy = pi[1] - pj[1], dz = pi[2] - pj[2];
  float d  = sqrtf(dx * dx + dy * dy + dz * dz + EPS_);
  float mm = mc[b * K_ + i] * mc[b * K_ + j];
  float* out = geo + (size_t)gid * BINS_;
#pragma unroll
  for (int tb = 0; tb < BINS_; ++tb) {
    float u = (d - tb * width) / width;
    out[tb] = __expf(-0.5f * u * u) * mm;
  }
}

// ---- MLP stage 1: h = LN(Zcat@W1 + b1); act = SiLU(h) (bf16 out) -----------
__global__ __launch_bounds__(256) void mlp_stage1(const float* __restrict__ sem,
                                                  const float* __restrict__ geo,
                                                  const float* __restrict__ W1,
                                                  const float* __restrict__ b1,
                                                  const float* __restrict__ lng,
                                                  const float* __restrict__ lnb,
                                                  __bf16* __restrict__ hact) {
  // W1 in fragment order: 5 k-tiles x 8 col-tiles, 528 halves per tile (42.2 KB)
  __shared__ __align__(32) __bf16 Wsw[5 * 8 * TSTR_];
  __shared__ float pb1[128], pg[128], pb[128];
  __shared__ __align__(16) unsigned char ubuf[32 * 132 * 4];  // 16.9 KB, reused
  __bf16* zc = reinterpret_cast<__bf16*>(ubuf);               // [32][168] row stage
  float*  hb = reinterpret_cast<float*>(ubuf);                // [32][132] h for LN
  const int t = threadIdx.x, lane = t & 31, w = t >> 5;

  for (int i = t; i < CINP_ * COUT_; i += 256) {   // W1 -> swizzled LDS (K zero-pad)
    int r = i >> 7, c = i & 127;
    float v = (r < CIN_) ? W1[r * COUT_ + c] : 0.f;
    int kt = r >> 5, kk = r & 31, ct = c >> 4, nn = c & 15;
    Wsw[(kt * 8 + ct) * TSTR_ + b_swz_idx(kk, nn)] = (__bf16)v;
  }
  if (t < 128) { pb1[t] = b1[t]; pg[t] = lng[t]; pb[t] = lnb[t]; }
  __syncthreads();

  const int rw = w >> 2, cw = w & 3;               // 2 row x 4 col waves
  const int NRB = (B_ * K_ * K_) / 32;             // 9216 row-tiles
  for (int rb = blockIdx.x; rb < NRB; rb += gridDim.x) {
    const size_t row0 = (size_t)rb * 32;
    __syncthreads();                               // ubuf reuse guard
    for (int i = t; i < 32 * 160; i += 256) {      // stage Zcat rows bf16
      int r = i / 160, c = i % 160;
      size_t gr = row0 + r;
      float v = (c < CZ_)  ? sem[gr * CZ_ + c]
              : (c < CIN_) ? geo[gr * BINS_ + (c - CZ_)]
                           : 0.f;
      zc[r * 168 + c] = (__bf16)v;
    }
    __syncthreads();
    v8f acc[2]; acc[0] = {}; acc[1] = {};
#pragma unroll
    for (int kt = 0; kt < CINP_ / 32; ++kt) {
      v16bf af = lds_a_16x32(zc + (rw * 16) * 168 + kt * 32, 168, lane);
#pragma unroll
      for (int j = 0; j < 2; ++j) {
        v16bf bf = lds_b_swz(Wsw + (kt * 8 + cw * 2 + j) * TSTR_, lane);
        acc[j] = __builtin_amdgcn_wmma_f32_16x16x32_bf16(false, af, false, bf,
                                                         (short)0, acc[j], false, false);
      }
    }
    __syncthreads();                               // done with zc -> hb
    const int mrow = (lane & 16) ? 8 : 0;
#pragma unroll
    for (int j = 0; j < 2; ++j) {
      int c = (cw * 2 + j) * 16 + (lane & 15);
      float bias = pb1[c];
#pragma unroll
      for (int v = 0; v < 8; ++v)
        hb[(rw * 16 + v + mrow) * 132 + c] = acc[j][v] + bias;
    }
    __syncthreads();
    {  // LayerNorm + SiLU: 8 threads per row, shfl-xor tree (width 8)
      int r = t >> 3, g = t & 7;
      float sum = 0.f, sq = 0.f;
#pragma unroll
      for (int i = 0; i < 16; ++i) {
        float x = hb[r * 132 + g * 16 + i];
        sum += x; sq += x * x;
      }
#pragma unroll
      for (int o = 4; o >= 1; o >>= 1) {
        sum += __shfl_xor(sum, o, 8);
        sq  += __shfl_xor(sq, o, 8);
      }
      float mu  = sum * (1.f / 128.f);
      float var = sq * (1.f / 128.f) - mu * mu;
      float rs  = rsqrtf(var + LN_EPS_);
      size_t gr = row0 + r;
#pragma unroll
      for (int i = 0; i < 16; ++i) {
        int c = g * 16 + i;
        float x = (hb[r * 132 + c] - mu) * rs * pg[c] + pb[c];
        hact[gr * COUT_ + c] = (__bf16)(x / (1.f + __expf(-x)));
      }
    }
  }
}

// ---- MLP stage 2: Z_c = (act @ W2 + b2) * mc2 ------------------------------
__global__ __launch_bounds__(256) void mlp_stage2(const __bf16* __restrict__ hact,
                                                  const float* __restrict__ W2,
                                                  const float* __restrict__ b2,
                                                  const float* __restrict__ mask_c,
                                                  float* __restrict__ outZ) {
  __shared__ __align__(32) __bf16 Wsw[4 * 8 * TSTR_];  // 33.8 KB, fragment order
  __shared__ __align__(32) __bf16 act[32 * 136];       // 8.7 KB, row-major A tile
  __shared__ float pb2[128];
  const int t = threadIdx.x, lane = t & 31, w = t >> 5;
  for (int i = t; i < COUT_ * COUT_; i += 256) {
    int r = i >> 7, c = i & 127;
    int kt = r >> 5, kk = r & 31, ct = c >> 4, nn = c & 15;
    Wsw[(kt * 8 + ct) * TSTR_ + b_swz_idx(kk, nn)] = (__bf16)W2[i];
  }
  if (t < 128) pb2[t] = b2[t];
  __syncthreads();
  const int rw = w >> 2, cw = w & 3;
  const int NRB = (B_ * K_ * K_) / 32;
  for (int rb = blockIdx.x; rb < NRB; rb += gridDim.x) {
    const size_t row0 = (size_t)rb * 32;
    __syncthreads();
    for (int i = t; i < 32 * 128; i += 256) {
      int r = i >> 7, c = i & 127;
      act[r * 136 + c] = hact[(row0 + r) * COUT_ + c];
    }
    __syncthreads();
    v8f acc[2]; acc[0] = {}; acc[1] = {};
#pragma unroll
    for (int kt = 0; kt < COUT_ / 32; ++kt) {
      v16bf af = lds_a_16x32(act + (rw * 16) * 136 + kt * 32, 136, lane);
#pragma unroll
      for (int j = 0; j < 2; ++j) {
        v16bf bf = lds_b_swz(Wsw + (kt * 8 + cw * 2 + j) * TSTR_, lane);
        acc[j] = __builtin_amdgcn_wmma_f32_16x16x32_bf16(false, af, false, bf,
                                                         (short)0, acc[j], false, false);
      }
    }
    const int mrow = (lane & 16) ? 8 : 0;
#pragma unroll
    for (int j = 0; j < 2; ++j) {
      int c = (cw * 2 + j) * 16 + (lane & 15);
      float bias = pb2[c];
#pragma unroll
      for (int v = 0; v < 8; ++v) {
        size_t gr  = row0 + (size_t)(rw * 16 + v + mrow);
        size_t rem = gr % ((size_t)K_ * K_);
        int bb = (int)(gr / ((size_t)K_ * K_));
        int ii = (int)(rem / K_), jj = (int)(rem % K_);
        float mm = mask_c[bb * K_ + ii] * mask_c[bb * K_ + jj];
        outZ[gr * COUT_ + c] = (acc[j][v] + bias) * mm;
      }
    }
  }
}

// ---------------------------------------------------------------------------
extern "C" void kernel_launch(void* const* d_in, const int* in_sizes, int n_in,
                              void* d_out, int out_size, void* d_ws, size_t ws_size,
                              hipStream_t stream) {
  const float* A      = (const float*)d_in[0];
  const float* Z_fine = (const float*)d_in[1];
  const float* mu_c   = (const float*)d_in[2];
  const float* mask_f = (const float*)d_in[3];
  const float* mask_c = (const float*)d_in[4];
  const float* W1     = (const float*)d_in[5];
  const float* b1     = (const float*)d_in[6];
  const float* ln_g   = (const float*)d_in[7];
  const float* ln_b   = (const float*)d_in[8];
  const float* W2     = (const float*)d_in[9];
  const float* b2     = (const float*)d_in[10];

  float* outZ    = (float*)d_out;                               // [B,K,K,128]
  float* out_sem = outZ + (size_t)B_ * K_ * K_ * COUT_;         // [B,K,N,128]
  float* out_geo = out_sem + (size_t)B_ * K_ * N_ * CZ_;        // [B,K,K,16]

  // Workspace layout (bytes):
  //   s:    B*N f32                    @ 0
  //   col:  B*K f32                    @ 4096
  //   AT:   B*K*N bf16                 @ 8192      (0.56 MB)
  //   ZbS:  B*12*3072*512 bf16 (swz)   @ 598016    (75.5 MB)
  //   hact: B*K*K*COUT bf16            @ 76095488  (75.5 MB)   total ~145 MB
  char* wsb = (char*)d_ws;
  float*  s_buf = (float*)(wsb + 0);
  float*  c_buf = (float*)(wsb + 4096);
  __bf16* AT    = (__bf16*)(wsb + 8192);
  __bf16* ZbS   = (__bf16*)(wsb + 598016);
  __bf16* hact  = (__bf16*)(wsb + 76095488ull);

  k_rowsum<<<dim3((B_ * N_ + 255) / 256), dim3(256), 0, stream>>>(A, mask_f, s_buf);
  k_colsum<<<dim3((B_ * K_ + 255) / 256), dim3(256), 0, stream>>>(A, mask_f, c_buf);
  k_at<<<dim3(K_ / 32, N_ / 32, B_), dim3(256), 0, stream>>>(A, mask_f, AT);
  k_zbf_swz<<<dim3(NCT_ / 8, NKT_, B_), dim3(256), 0, stream>>>(Z_fine, mask_f, ZbS);

  gemm1_sem<<<dim3(NCOL_ / 128, K_ / 64, B_), dim3(256), 0, stream>>>(
      AT, ZbS, s_buf, c_buf, out_sem);

  k_rbf<<<dim3((B_ * K_ * K_ + 255) / 256), dim3(256), 0, stream>>>(mu_c, mask_c, out_geo);

  mlp_stage1<<<dim3(2304), dim3(256), 0, stream>>>(out_sem, out_geo, W1, b1,
                                                   ln_g, ln_b, hact);
  mlp_stage2<<<dim3(2304), dim3(256), 0, stream>>>(hact, W2, b2, mask_c, outZ);
}
